// RegionProposalNetwork_26731876451105
// MI455X (gfx1250) — matrix operand order repaired and checked by reading, not compiled, not run on hardware
//
#include <hip/hip_runtime.h>
#include <stdint.h>

// ---------------- problem constants (match reference) ----------------
#define C_IN   256
#define Hh     128
#define Ww     128
#define HW     16384          // 128*128
#define KCONV  2304           // 9*256
#define NANCH  147456         // HW*9
#define KSEL   6000
#define CAP    8192           // sort capacity (pow2 >= KSEL)
#define NOUT   1000
#define NMSW   188            // ceil(6000/32)
#define IOU_T  0.7f
#define CLIPV  4.135166556742356f   // log(1000/16)

// ---------------- WMMA types (gfx1250 CDNA5) ----------------
typedef __attribute__((ext_vector_type(16))) __bf16 v16bf;
typedef __attribute__((ext_vector_type(8)))  float  v8f;
typedef __attribute__((ext_vector_type(4)))  int    v4i;

struct __attribute__((aligned(16))) U128 { unsigned x, y, z, w; };
union  ABu  { v16bf v; U128 q[2]; };

__device__ __forceinline__ unsigned short f2bf(float f) {
  unsigned u = __float_as_uint(f);
  u += 0x7FFFu + ((u >> 16) & 1u);        // round-to-nearest-even
  return (unsigned short)(u >> 16);
}

__device__ __forceinline__ unsigned orderable(float s) {
  unsigned u = __float_as_uint(s);
  return u ^ ((u >> 31) ? 0xFFFFFFFFu : 0x80000000u);
}

// ---------------- CDNA5 async global->LDS copy (ASYNCcnt path) ----------------
// clang prototype: (v4i addrspace(1)* src, v4i addrspace(3)* dst, const int off, const int cpol)
__device__ __forceinline__ void async_b128(void* lptr, const void* gptr) {
#if defined(__has_builtin) && __has_builtin(__builtin_amdgcn_global_load_async_to_lds_b128)
  __builtin_amdgcn_global_load_async_to_lds_b128(
      (__attribute__((address_space(1))) v4i*)(size_t)gptr,
      (__attribute__((address_space(3))) v4i*)lptr, 0, 0);
#else
  unsigned loff = (unsigned)(size_t)(__attribute__((address_space(3))) void*)lptr;
  unsigned long long ga = (unsigned long long)(size_t)gptr;
  asm volatile("global_load_async_to_lds_b128 %0, %1, off"
               :: "v"(loff), "v"(ga) : "memory");
#endif
}

__device__ __forceinline__ void wait_async() {
#if defined(__has_builtin) && __has_builtin(__builtin_amdgcn_s_wait_asynccnt)
  __builtin_amdgcn_s_wait_asynccnt(0);
#else
  asm volatile("s_wait_asynccnt 0x0" ::: "memory");
#endif
}

// ---------------- misc: zero a u32 span ----------------
__global__ void k_zero(unsigned* p, int n) {
  int i = blockIdx.x * 256 + threadIdx.x;
  if (i < n) p[i] = 0u;
}

// ---------------- pack conv weights into WMMA B layout ----------------
// W2[o][tap*256 + c] = bf16(conv_w[o][c][ky][kx]), o-major rows (k contiguous)
__global__ void k_reorder_w(const float* __restrict__ w, unsigned short* __restrict__ W2) {
  int i = blockIdx.x * 256 + threadIdx.x;       // < 256*2304
  if (i >= C_IN * KCONV) return;
  int o = i / KCONV, r = i - o * KCONV;         // r = tap*256 + c
  int tap = r >> 8, c = r & 255;
  W2[i] = f2bf(w[o * KCONV + c * 9 + tap]);
}

// W3[n][c], n: 0..8 cls, 9..44 bbox, 45..47 zero pad
__global__ void k_reorder_w3(const float* __restrict__ cls_w, const float* __restrict__ bbox_w,
                             unsigned short* __restrict__ W3) {
  int i = blockIdx.x * 256 + threadIdx.x;       // < 48*256
  if (i >= 48 * 256) return;
  int n = i >> 8, c = i & 255;
  float v = 0.f;
  if (n < 9)       v = cls_w[n * 256 + c];
  else if (n < 45) v = bbox_w[(n - 9) * 256 + c];
  W3[i] = f2bf(v);
}

// ---------------- 3x3 conv + bias + ReLU as implicit GEMM (bf16 WMMA) ----------------
// grid (4 x-tiles of 32, 128 rows, 4 n64-tiles), 256 threads = 8 waves.
// Block tile: 32 pixels x 64 out-channels. A staged via LDS (f32->bf16 convert,
// block-uniform interior fast path); B panel (64x256 per tap, 32 KB) double-buffered
// in LDS via async global->LDS copies overlapped with compute.
__global__ void __launch_bounds__(256)
k_conv3x3(const float* __restrict__ feat, const unsigned short* __restrict__ W2,
          const float* __restrict__ bias, unsigned short* __restrict__ t) {
  extern __shared__ char smem[];
  unsigned short* At = (unsigned short*)smem;              // [32 px][32 ch] bf16, 4 KB
  unsigned short* Bb = (unsigned short*)(smem + 4096);     // 2 x [64][256] bf16, 64 KB

  const int tid  = threadIdx.x;
  const int lane = tid & 31, wave = tid >> 5;
  const int mw   = wave & 1;           // pixel half (16 rows)
  const int nw   = wave >> 1;          // 16-column group
  const int x0   = blockIdx.x << 5;
  const int y    = blockIdx.y;
  const int nblk = blockIdx.z << 6;
  const int mrow = lane & 15;
  const int half = (lane < 16) ? 0 : 1;

  auto stageB = [&](int tap, int buf) {       // 2048 x 16B units, 8 per thread
    unsigned short* dst = Bb + buf * (64 * 256);
#pragma unroll
    for (int k = 0; k < 8; ++k) {
      int u = tid + (k << 8);
      int ncol = u >> 5, seg = u & 31;
      async_b128(dst + ncol * 256 + seg * 8,
                 W2 + (size_t)(nblk + ncol) * KCONV + tap * 256 + seg * 8);
    }
  };
  stageB(0, 0);

  v8f acc = {};

  for (int tap = 0; tap < 9; ++tap) {
    wait_async();
    __syncthreads();                          // B panel (tap&1) visible to all waves
    if (tap < 8) stageB(tap + 1, (tap + 1) & 1);
    const unsigned short* Bcur = Bb + (tap & 1) * (64 * 256);

    const int dy = tap / 3 - 1, dx = tap % 3 - 1;
    const int yy = y + dy;
    const bool yok = ((unsigned)yy < (unsigned)Hh);
    const int  xs  = x0 + dx;
    const bool interior = yok && (xs >= 0) && (xs + 31 < Ww);

    for (int c0 = 0; c0 < C_IN; c0 += 32) {
      __syncthreads();                        // previous chunk's A reads done
      if (interior) {                         // fast path: no predicates
        int c  = tid >> 3;
        int mg = (tid & 7) << 2;
        const float* src = feat + (size_t)(c0 + c) * HW + yy * Ww + xs + mg;
        float v0 = src[0], v1 = src[1], v2 = src[2], v3 = src[3];
        __builtin_prefetch(src + 32 * HW, 0, 1);   // next K-chunk, same line
        At[(mg + 0) * 32 + c] = f2bf(v0);
        At[(mg + 1) * 32 + c] = f2bf(v1);
        At[(mg + 2) * 32 + c] = f2bf(v2);
        At[(mg + 3) * 32 + c] = f2bf(v3);
      } else {                                // boundary tiles only
        for (int e = tid; e < 1024; e += 256) {
          int m = e >> 5, c = e & 31;
          int xx = xs + m;
          float v = 0.f;
          if (yok && (unsigned)xx < (unsigned)Ww)
            v = feat[(size_t)(c0 + c) * HW + yy * Ww + xx];
          At[m * 32 + c] = f2bf(v);
        }
      }
      __syncthreads();

      ABu a, b;
      // A operand: lanes<16 -> K 0..15 ; lanes>=16 -> K 16..31 (contiguous, matches B)
      const U128* ap = reinterpret_cast<const U128*>(&At[(mw * 16 + mrow) * 32 + half * 16]);
      a.q[0] = ap[0];
      a.q[1] = ap[1];
      // B operand: lane holds column, 16 contiguous K values per lane half
      const U128* bq = reinterpret_cast<const U128*>(
          Bcur + (size_t)(nw * 16 + mrow) * 256 + c0 + half * 16);
      b.q[0] = bq[0]; b.q[1] = bq[1];

      acc = __builtin_amdgcn_wmma_f32_16x16x32_bf16(false, a.v, false, b.v,
                                                    (short)0, acc, false, false);
    }
  }
  const int   ncol = nblk + nw * 16 + mrow;
  const float bv   = bias[ncol];
#pragma unroll
  for (int v = 0; v < 8; ++v) {
    int   m = mw * 16 + v + half * 8;         // C/D layout: VGPR v -> M=v (+8 upper lanes)
    float r = acc[v] + bv;
    r = r > 0.f ? r : 0.f;
    t[(size_t)(y * Ww + x0 + m) * 256 + ncol] = f2bf(r);
  }
}

// ---------------- 1x1 heads (cls 9 + bbox 36, padded to 48) via WMMA ----------------
// grid (1024 pixel-tiles, 3 n16-tiles), 32 threads (one wave).
__global__ void __launch_bounds__(32)
k_heads(const unsigned short* __restrict__ t, const unsigned short* __restrict__ W3,
        const float* __restrict__ cls_b, const float* __restrict__ bbox_b,
        float* __restrict__ scores, float* __restrict__ deltas) {
  const int lane = threadIdx.x;
  const int p0   = blockIdx.x << 4;
  const int n0   = blockIdx.y << 4;
  const int mrow = lane & 15;
  const int half = (lane < 16) ? 0 : 1;
  v8f acc = {};
  for (int c0 = 0; c0 < 256; c0 += 32) {
    ABu a, b;
    const U128* ap = reinterpret_cast<const U128*>(t + (size_t)(p0 + mrow) * 256 + c0 + half * 16);
    a.q[0] = ap[0]; a.q[1] = ap[1];
    const U128* bq = reinterpret_cast<const U128*>(W3 + (size_t)(n0 + mrow) * 256 + c0 + half * 16);
    b.q[0] = bq[0]; b.q[1] = bq[1];
    acc = __builtin_amdgcn_wmma_f32_16x16x32_bf16(false, a.v, false, b.v,
                                                  (short)0, acc, false, false);
  }
  const int n = n0 + mrow;
  float bv = 0.f;
  if (n < 9) bv = cls_b[n]; else if (n < 45) bv = bbox_b[n - 9];
#pragma unroll
  for (int v = 0; v < 8; ++v) {
    const int   p = p0 + v + half * 8;
    const float r = acc[v] + bv;
    if (n < 9) scores[(size_t)p * 9 + n] = r;                 // (H,W,A) order
    else if (n < 45) {
      int a9 = (n - 9) >> 2, d = (n - 9) & 3;
      deltas[((size_t)p * 9 + a9) * 4 + d] = r;
    }
  }
}

// ---------------- top-k via bit-histogram threshold + compact ----------------
__global__ void k_hist(const float* __restrict__ scores, unsigned* __restrict__ hist) {
  int i = blockIdx.x * 256 + threadIdx.x;
  if (i >= NANCH) return;
  atomicAdd(&hist[orderable(scores[i]) >> 21], 1u);           // 2048 bins
}

__global__ void k_thresh(const unsigned* __restrict__ hist, unsigned* __restrict__ params) {
  if (threadIdx.x != 0) return;
  unsigned cum = 0; int tb = 0;
  for (int b = 2047; b >= 0; --b) {
    cum += hist[b];
    if (cum >= KSEL) { tb = b; break; }
  }
  params[0] = (unsigned)tb;
}

__global__ void k_compact(const float* __restrict__ scores, const unsigned* __restrict__ params,
                          unsigned* __restrict__ counter, unsigned long long* __restrict__ pairs) {
  int i = blockIdx.x * 256 + threadIdx.x;
  if (i >= NANCH) return;
  unsigned u = orderable(scores[i]);
  if ((u >> 21) >= params[0]) {
    unsigned pos = atomicAdd(counter, 1u);
    if (pos < CAP)                                            // key ties broken by ~idx
      pairs[pos] = ((unsigned long long)u << 32) |
                   (unsigned long long)(0xFFFFFFFFu - (unsigned)i);
  }
}

// one-workgroup bitonic sort of 8192 u64 keys, descending (64 KB dynamic LDS)
__global__ void __launch_bounds__(1024)
k_sort(const unsigned long long* __restrict__ pairs, unsigned* __restrict__ top_idx) {
  extern __shared__ unsigned long long sp[];
  const int tid = threadIdx.x;
  for (int e = tid; e < CAP; e += 1024) sp[e] = pairs[e];
  __syncthreads();
  for (int k = 2; k <= CAP; k <<= 1) {
    for (int j = k >> 1; j > 0; j >>= 1) {
      for (int e = tid; e < CAP; e += 1024) {
        int p = e ^ j;
        if (p > e) {
          unsigned long long a = sp[e], b = sp[p];
          bool desc = ((e & k) == 0);
          if ((a < b) == desc) { sp[e] = b; sp[p] = a; }
        }
      }
      __syncthreads();
    }
  }
  for (int i = tid; i < KSEL; i += 1024)
    top_idx[i] = 0xFFFFFFFFu - (unsigned)(sp[i] & 0xFFFFFFFFull);
}

// ---------------- box decode + clip + min-size ----------------
__global__ void k_decode(const unsigned* __restrict__ top_idx, const float* __restrict__ deltas,
                         const float* __restrict__ anchors, const int* __restrict__ img_h,
                         const int* __restrict__ img_w, float* __restrict__ boxes,
                         int* __restrict__ valid) {
  int i = blockIdx.x * 256 + threadIdx.x;
  if (i >= KSEL) return;
  unsigned idx = top_idx[i];
  float bx0 = 0.f, bx1 = 0.f, bx2 = 0.f, bx3 = 0.f; int ok = 0;
  if (idx < (unsigned)NANCH) {
    const float* d = deltas  + (size_t)idx * 4;
    const float* a = anchors + (size_t)idx * 4;
    float aw = a[2] - a[0], ah = a[3] - a[1];
    float ax = a[0] + 0.5f * aw, ay = a[1] + 0.5f * ah;
    float dw = fminf(d[2], CLIPV), dh = fminf(d[3], CLIPV);
    float px = d[0] * aw + ax, py = d[1] * ah + ay;
    float pw = __expf(dw) * aw, ph = __expf(dh) * ah;
    float fw = (float)img_w[0], fh = (float)img_h[0];
    bx0 = fminf(fmaxf(px - 0.5f * pw, 0.f), fw);
    bx1 = fminf(fmaxf(py - 0.5f * ph, 0.f), fh);
    bx2 = fminf(fmaxf(px + 0.5f * pw, 0.f), fw);
    bx3 = fminf(fmaxf(py + 0.5f * ph, 0.f), fh);
    ok = (bx2 - bx0 >= 1.0f) && (bx3 - bx1 >= 1.0f);
  }
  boxes[i * 4 + 0] = bx0; boxes[i * 4 + 1] = bx1;
  boxes[i * 4 + 2] = bx2; boxes[i * 4 + 3] = bx3;
  valid[i] = ok;
}

// ---------------- NMS: suppression bitmask (cols > i only) ----------------
__global__ void __launch_bounds__(64)
k_nms_mask(const float* __restrict__ boxes, unsigned* __restrict__ mask) {
  const int i = blockIdx.y;
  const int j = blockIdx.x * 64 + threadIdx.x;
  if (j >= NMSW) return;
  const float ax1 = boxes[i * 4 + 0], ay1 = boxes[i * 4 + 1];
  const float ax2 = boxes[i * 4 + 2], ay2 = boxes[i * 4 + 3];
  const float areaA = (ax2 - ax1) * (ay2 - ay1);
  unsigned bits = 0;
  const int c0 = j << 5;
#pragma unroll 4
  for (int b = 0; b < 32; ++b) {
    int c = c0 + b;
    if (c >= KSEL || c <= i) continue;
    float bx1 = boxes[c * 4 + 0], by1 = boxes[c * 4 + 1];
    float bx2 = boxes[c * 4 + 2], by2 = boxes[c * 4 + 3];
    float areaB = (bx2 - bx1) * (by2 - by1);
    float ix = fmaxf(fminf(ax2, bx2) - fmaxf(ax1, bx1), 0.f);
    float iy = fmaxf(fminf(ay2, by2) - fmaxf(ay1, by1), 0.f);
    float inter = ix * iy;
    float iou = inter / (areaA + areaB - inter);
    if (iou > IOU_T) bits |= (1u << b);
  }
  mask[(size_t)i * NMSW + j] = bits;
}

// ---------------- NMS: sequential greedy scan + emit first 1000 ----------------
__global__ void __launch_bounds__(256)
k_nms_scan(const unsigned* __restrict__ mask, const int* __restrict__ valid,
           const float* __restrict__ boxes, float* __restrict__ out) {
  __shared__ unsigned keep[NMSW];
  __shared__ int kflag, rank, stop;
  const int tid = threadIdx.x;
  for (int e = tid; e < NOUT * 4; e += 256) out[e] = 0.f;
  if (tid < NMSW) {
    unsigned w = 0;
    for (int b = 0; b < 32; ++b) {
      int c = (tid << 5) + b;
      if (c < KSEL && valid[c]) w |= (1u << b);
    }
    keep[tid] = w;
  }
  if (tid == 0) { rank = 0; stop = 0; }
  __syncthreads();
  for (int i = 0; i < KSEL; ++i) {
    if (tid == 0) kflag = (keep[i >> 5] >> (i & 31)) & 1;
    __syncthreads();
    if (kflag) {
      for (int j = tid; j < NMSW; j += 256) keep[j] &= ~mask[(size_t)i * NMSW + j];
      if (tid == 0) {
        if (rank < NOUT) {
          out[rank * 4 + 0] = boxes[i * 4 + 0];
          out[rank * 4 + 1] = boxes[i * 4 + 1];
          out[rank * 4 + 2] = boxes[i * 4 + 2];
          out[rank * 4 + 3] = boxes[i * 4 + 3];
        }
        rank++;
        if (rank >= NOUT) stop = 1;             // later kept boxes can't affect output
      }
    }
    __syncthreads();
    if (stop) break;
  }
}

// ---------------- host launcher ----------------
extern "C" void kernel_launch(void* const* d_in, const int* in_sizes, int n_in,
                              void* d_out, int out_size, void* d_ws, size_t ws_size,
                              hipStream_t stream) {
  (void)in_sizes; (void)n_in; (void)out_size; (void)ws_size;
  const float* feature = (const float*)d_in[0];
  const float* conv_w  = (const float*)d_in[1];
  const float* conv_b  = (const float*)d_in[2];
  const float* cls_w   = (const float*)d_in[3];
  const float* cls_b   = (const float*)d_in[4];
  const float* bbox_w  = (const float*)d_in[5];
  const float* bbox_b  = (const float*)d_in[6];
  const float* anchors = (const float*)d_in[7];
  const int*   img_h   = (const int*)d_in[8];
  const int*   img_w   = (const int*)d_in[9];
  float* out = (float*)d_out;

  // workspace layout (~17.3 MB total)
  char* ws = (char*)d_ws;
  size_t off = 0;
  auto take = [&](size_t bytes) -> void* {
    void* p = ws + off;
    off = (off + bytes + 255) & ~(size_t)255;
    return p;
  };
  unsigned short*      W2      = (unsigned short*)take((size_t)C_IN * KCONV * 2); // 1.18 MB
  unsigned short*      W3      = (unsigned short*)take(48 * 256 * 2);
  unsigned short*      t_act   = (unsigned short*)take((size_t)HW * 256 * 2);     // 8.4 MB
  float*               scores  = (float*)take((size_t)NANCH * 4);                 // 0.59 MB
  float*               deltas  = (float*)take((size_t)NANCH * 4 * 4);             // 2.36 MB
  unsigned*            hist    = (unsigned*)take(2048 * 4);
  unsigned*            params  = (unsigned*)take(64);
  unsigned*            counter = (unsigned*)take(64);
  unsigned long long*  pairs   = (unsigned long long*)take((size_t)CAP * 8);      // 64 KB
  unsigned*            top_idx = (unsigned*)take((size_t)KSEL * 4);
  float*               boxes   = (float*)take((size_t)KSEL * 4 * 4);
  int*                 valid   = (int*)take((size_t)KSEL * 4);
  unsigned*            nmsmask = (unsigned*)take((size_t)KSEL * NMSW * 4);        // 4.5 MB

  // per-launch zero init of accumulated state
  k_zero<<<(2048 + 255) / 256, 256, 0, stream>>>(hist, 2048);
  k_zero<<<1, 256, 0, stream>>>(params, 16);
  k_zero<<<1, 256, 0, stream>>>(counter, 16);
  k_zero<<<(CAP * 2 + 255) / 256, 256, 0, stream>>>((unsigned*)pairs, CAP * 2);

  // weight packing (once per launch; cheap)
  k_reorder_w<<<(C_IN * KCONV + 255) / 256, 256, 0, stream>>>(conv_w, W2);
  k_reorder_w3<<<48, 256, 0, stream>>>(cls_w, bbox_w, W3);

  // conv backbone + heads (WMMA); conv uses 4KB A-tile + 64KB double-buffered B panel
  k_conv3x3<<<dim3(Ww / 32, Hh, 4), 256, 69632, stream>>>(feature, W2, conv_b, t_act);
  k_heads<<<dim3(HW / 16, 3), 32, 0, stream>>>(t_act, W3, cls_b, bbox_b, scores, deltas);

  // top-6000 selection + deterministic sort
  k_hist<<<NANCH / 256, 256, 0, stream>>>(scores, hist);
  k_thresh<<<1, 32, 0, stream>>>(hist, params);
  k_compact<<<NANCH / 256, 256, 0, stream>>>(scores, params, counter, pairs);
  k_sort<<<1, 1024, (size_t)CAP * 8, stream>>>(pairs, top_idx);

  // decode + NMS
  k_decode<<<(KSEL + 255) / 256, 256, 0, stream>>>(top_idx, deltas, anchors, img_h, img_w,
                                                   boxes, valid);
  k_nms_mask<<<dim3((NMSW + 63) / 64, KSEL), 64, 0, stream>>>(boxes, nmsmask);
  k_nms_scan<<<1, 256, 0, stream>>>(nmsmask, valid, boxes, out);
}
